// HEPOSBartAttention_8057358648078
// MI455X (gfx1250) — matrix operand order repaired and checked by reading, not compiled, not run on hardware
//
#include <hip/hip_runtime.h>
#include <hip/hip_bf16.h>

typedef __bf16 bf16_t;
typedef __attribute__((ext_vector_type(16))) __bf16 v16bf;
typedef __attribute__((ext_vector_type(8)))  __bf16 v8bf;
typedef __attribute__((ext_vector_type(4)))  __bf16 v4bf;
typedef __attribute__((ext_vector_type(8)))  float  v8f;
typedef __attribute__((ext_vector_type(4)))  float  v4f;
typedef __attribute__((ext_vector_type(4)))  int    v4i;

#define WMMA_BF16(A, B, C) \
  __builtin_amdgcn_wmma_f32_16x16x32_bf16(false, (A), false, (B), (short)0, (C), false, false)

// ---------------- fragment loaders (wave32; contiguous k per lane half, same order for A and B) ----------------
// A-matrix 16x32 bf16: lane L holds row M=L%16; elements e -> K=k0+(L/16)*16+e.
// Two adjacent 16B chunks.
__device__ __forceinline__ v16bf load_afrag(const bf16_t* __restrict__ base, size_t rowstride,
                                            int row0, int k0, int lane) {
  int r = row0 + (lane & 15);
  int g = lane >> 4;
  const bf16_t* p = base + (size_t)r * rowstride + k0 + g * 16;
  v8bf lo = *(const v8bf*)(p);
  v8bf hi = *(const v8bf*)(p + 8);
  return __builtin_shufflevector(lo, hi, 0,1,2,3,4,5,6,7,8,9,10,11,12,13,14,15);
}

// A-fragment sourced from fp32 data (convert on the fly)
__device__ __forceinline__ v16bf load_afrag_f32(const float* __restrict__ base, size_t rowstride,
                                                int row0, int k0, int lane) {
  int r = row0 + (lane & 15);
  int g = lane >> 4;
  const float* p = base + (size_t)r * rowstride + k0 + g * 16;
  v16bf out;
#pragma unroll
  for (int e = 0; e < 8; ++e) {
    out[e]     = (bf16_t)p[e];
    out[e + 8] = (bf16_t)p[e + 8];
  }
  return out;
}

// B-matrix 32x16 bf16 representing W^T (B[k][n] = W[n][k]): lane L holds col N=n0+L%16,
// elements e -> K = k0 + (L/16)*16 + e. One contiguous 32B chunk from row n of W.
__device__ __forceinline__ v16bf load_bfragT(const bf16_t* __restrict__ base, size_t rowstride,
                                             int n0, int k0, int lane) {
  int n = n0 + (lane & 15);
  int g = lane >> 4;
  const bf16_t* p = base + (size_t)n * rowstride + k0 + g * 16;
  return *(const v16bf*)p;
}

__device__ __forceinline__ v16bf load_bfragT_f32(const float* __restrict__ base, size_t rowstride,
                                                 int n0, int k0, int lane) {
  int n = n0 + (lane & 15);
  int g = lane >> 4;
  const float* p = base + (size_t)n * rowstride + k0 + g * 16;
  v16bf out;
#pragma unroll
  for (int e = 0; e < 16; ++e) out[e] = (bf16_t)p[e];
  return out;
}

// ---------------- kernel 0: fp32 -> bf16 convert ----------------
__global__ __launch_bounds__(256) void cvt_f32_bf16_kernel(const float* __restrict__ in,
                                                           bf16_t* __restrict__ out, int n4) {
  int i = blockIdx.x * blockDim.x + threadIdx.x;
  if (i < n4) {
    v4f x = *(const v4f*)(in + (size_t)i * 4);
    v4bf o;
#pragma unroll
    for (int e = 0; e < 4; ++e) o[e] = (bf16_t)x[e];
    *(v4bf*)(out + (size_t)i * 4) = o;
  }
}

// ---------------- kernel 1/4: out[i][j] = sum_k X[i][k]*W[j][k] + bias[j] ----------------
// WMMA-M = feature j, WMMA-N = token i (transposed tile -> all contiguous loads/stores).
// Ping-pong double buffering (manual 2x unroll): no register rotation, loads for the next
// 32-k half are in flight while the current half's WMMAs execute.
__global__ __launch_bounds__(128) void gemm_xwt_kernel(const bf16_t* __restrict__ X,
                                                       const bf16_t* __restrict__ W,
                                                       const float* __restrict__ bias,
                                                       bf16_t* __restrict__ outB,
                                                       float* __restrict__ outF,
                                                       int N, int K) {
  int lane = threadIdx.x & 31;
  int wave = threadIdx.x >> 5;
  int i0 = blockIdx.x * 16;
  int j0 = blockIdx.y * 256 + wave * 64;
  v8f acc[4] = {};

  v16bf bf0 = load_bfragT(X, K, i0, 0, lane);
  v16bf af0[4];
#pragma unroll
  for (int t = 0; t < 4; ++t) af0[t] = load_afrag(W, K, j0 + t * 16, 0, lane);

  for (int k0 = 0; k0 < K; k0 += 64) {
    int k1 = k0 + 32;
    v16bf bf1 = load_bfragT(X, K, i0, k1, lane);
    v16bf af1[4];
#pragma unroll
    for (int t = 0; t < 4; ++t) af1[t] = load_afrag(W, K, j0 + t * 16, k1, lane);
#pragma unroll
    for (int t = 0; t < 4; ++t) acc[t] = WMMA_BF16(af0[t], bf0, acc[t]);

    int k2 = (k0 + 64 < K) ? (k0 + 64) : 0;  // wrap: harmless redundant prefetch at end
    bf0 = load_bfragT(X, K, i0, k2, lane);
#pragma unroll
    for (int t = 0; t < 4; ++t) af0[t] = load_afrag(W, K, j0 + t * 16, k2, lane);
#pragma unroll
    for (int t = 0; t < 4; ++t) acc[t] = WMMA_BF16(af1[t], bf1, acc[t]);
  }

  int g = lane >> 4;
  int iq = i0 + (lane & 15);
#pragma unroll
  for (int t = 0; t < 4; ++t) {
    int jb = j0 + t * 16 + g * 8;
    const float* bp = bias + jb;
#pragma unroll
    for (int r = 0; r < 8; ++r) acc[t][r] += bp[r];
    if (outB) {
      v8bf ob;
#pragma unroll
      for (int r = 0; r < 8; ++r) ob[r] = (bf16_t)acc[t][r];
      *(v8bf*)(outB + (size_t)iq * N + jb) = ob;
    } else {
      *(v8f*)(outF + (size_t)iq * N + jb) = acc[t];
    }
  }
}

// ---------------- kernel 2: gather-aware K/V projection ----------------
// Head h uses kv tokens t = h + 16*j (j = 0..511). Each kv element is read by five waves
// (one K wave and each of the 4 V waves), converted fp32->bf16 in-register.
// Kg[b][h][j][64] row-major; VgT[b][h][64][512] transposed. Ping-pong k-loop.
__global__ __launch_bounds__(256) void kv_gather_proj_kernel(const float* __restrict__ KV,
                                                             const bf16_t* __restrict__ Wk,
                                                             const float* __restrict__ bk,
                                                             const bf16_t* __restrict__ Wv,
                                                             const float* __restrict__ bv,
                                                             bf16_t* __restrict__ Kg,
                                                             bf16_t* __restrict__ VgT) {
  const int E = 1024, Tk = 8192, Mg = 512;
  int lane = threadIdx.x & 31;
  int wave = threadIdx.x >> 5;
  int b = blockIdx.z, h = blockIdx.y;
  int jc0 = blockIdx.x * 64;
  const float* xbase = KV + ((size_t)b * Tk + h) * E;  // row j -> token h+16j
  const size_t xstride = (size_t)16 * E;
  int g = lane >> 4;

  if (wave < 4) {
    // K: WMMA-M = d (4 tiles = 64), WMMA-N = j (16 per wave)
    int j0 = jc0 + wave * 16;
    v8f acc[4] = {};
    v16bf bf0 = load_bfragT_f32(xbase, xstride, j0, 0, lane);
    v16bf af0[4];
#pragma unroll
    for (int t = 0; t < 4; ++t) af0[t] = load_afrag(Wk, E, h * 64 + t * 16, 0, lane);
    for (int k0 = 0; k0 < E; k0 += 64) {
      int k1 = k0 + 32;
      v16bf bf1 = load_bfragT_f32(xbase, xstride, j0, k1, lane);
      v16bf af1[4];
#pragma unroll
      for (int t = 0; t < 4; ++t) af1[t] = load_afrag(Wk, E, h * 64 + t * 16, k1, lane);
#pragma unroll
      for (int t = 0; t < 4; ++t) acc[t] = WMMA_BF16(af0[t], bf0, acc[t]);
      int k2 = (k0 + 64 < E) ? (k0 + 64) : 0;
      bf0 = load_bfragT_f32(xbase, xstride, j0, k2, lane);
#pragma unroll
      for (int t = 0; t < 4; ++t) af0[t] = load_afrag(Wk, E, h * 64 + t * 16, k2, lane);
#pragma unroll
      for (int t = 0; t < 4; ++t) acc[t] = WMMA_BF16(af1[t], bf1, acc[t]);
    }
    int j = j0 + (lane & 15);
    bf16_t* kout = Kg + (((size_t)b * 16 + h) * Mg + j) * 64;
#pragma unroll
    for (int t = 0; t < 4; ++t) {
      int db = t * 16 + g * 8;
      v8bf ob;
#pragma unroll
      for (int r = 0; r < 8; ++r) ob[r] = (bf16_t)(acc[t][r] + bk[h * 64 + db + r]);
      *(v8bf*)(kout + db) = ob;
    }
  } else {
    // V transposed: WMMA-M = j (4 tiles = 64), WMMA-N = d (16 per wave)
    int d0 = (wave - 4) * 16;
    float bvv = bv[h * 64 + d0 + (lane & 15)];
    v8f acc[4] = {};
    v16bf bf0 = load_bfragT(Wv, E, h * 64 + d0, 0, lane);
    v16bf af0[4];
#pragma unroll
    for (int t = 0; t < 4; ++t) af0[t] = load_afrag_f32(xbase, xstride, jc0 + t * 16, 0, lane);
    for (int k0 = 0; k0 < E; k0 += 64) {
      int k1 = k0 + 32;
      v16bf bf1 = load_bfragT(Wv, E, h * 64 + d0, k1, lane);
      v16bf af1[4];
#pragma unroll
      for (int t = 0; t < 4; ++t) af1[t] = load_afrag_f32(xbase, xstride, jc0 + t * 16, k1, lane);
#pragma unroll
      for (int t = 0; t < 4; ++t) acc[t] = WMMA_BF16(af0[t], bf0, acc[t]);
      int k2 = (k0 + 64 < E) ? (k0 + 64) : 0;
      bf0 = load_bfragT(Wv, E, h * 64 + d0, k2, lane);
#pragma unroll
      for (int t = 0; t < 4; ++t) af0[t] = load_afrag_f32(xbase, xstride, jc0 + t * 16, k2, lane);
#pragma unroll
      for (int t = 0; t < 4; ++t) acc[t] = WMMA_BF16(af1[t], bf1, acc[t]);
    }
    int d = d0 + (lane & 15);
    bf16_t* vout = VgT + (((size_t)b * 16 + h) * 64 + d) * Mg;
#pragma unroll
    for (int t = 0; t < 4; ++t) {
      int jb = jc0 + t * 16 + g * 8;
      v8bf ob;
#pragma unroll
      for (int r = 0; r < 8; ++r) ob[r] = (bf16_t)(acc[t][r] + bvv);
      *(v8bf*)(vout + jb) = ob;
    }
  }
}

// ---------------- kernel 3: flash-style attention per (b,h) ----------------
// Wave owns 16 queries. Computes S^T = Kg . Q^T so softmax rows live along register dim.
// Chunk body as macro; kfA/kfB ping-pong across pairs of 32-key chunks (no rotation moves).
// KF: current chunk's Kg fragments (read); KFN: next chunk's (written by prefetch).
#define ATTN_CHUNK(KF, KFN, JC, JN)                                            \
  {                                                                            \
    v8f s0 = {}, s1 = {};                                                      \
    s0 = WMMA_BF16(KF[0], qf0, s0);                                            \
    s0 = WMMA_BF16(KF[1], qf1, s0);                                            \
    s1 = WMMA_BF16(KF[2], qf0, s1);                                            \
    s1 = WMMA_BF16(KF[3], qf1, s1);                                            \
    v16bf vf0 = load_afrag(vbase, Mg, 0,  (JC), lane);                         \
    v16bf vf1 = load_afrag(vbase, Mg, 16, (JC), lane);                         \
    v16bf vf2 = load_afrag(vbase, Mg, 32, (JC), lane);                         \
    v16bf vf3 = load_afrag(vbase, Mg, 48, (JC), lane);                         \
    KFN[0] = load_afrag(kbase, 64, (JN), 0, lane);                             \
    KFN[1] = load_afrag(kbase, 64, (JN), 32, lane);                            \
    KFN[2] = load_afrag(kbase, 64, (JN) + 16, 0, lane);                        \
    KFN[3] = load_afrag(kbase, 64, (JN) + 16, 32, lane);                       \
    float mloc = m_run;                                                        \
    _Pragma("unroll") for (int r = 0; r < 8; ++r) {                            \
      s0[r] *= scaling;                                                        \
      s1[r] *= scaling;                                                        \
      mloc = fmaxf(mloc, fmaxf(s0[r], s1[r]));                                 \
    }                                                                          \
    mloc = fmaxf(mloc, __shfl_xor(mloc, 16, 32));                              \
    float corr = __expf(m_run - mloc);                                         \
    m_run = mloc;                                                              \
    l_run *= corr;                                                             \
    v8bf own0, own1;                                                           \
    _Pragma("unroll") for (int r = 0; r < 8; ++r) {                            \
      float p0 = __expf(s0[r] - mloc);                                         \
      float p1 = __expf(s1[r] - mloc);                                         \
      l_run += p0 + p1;                                                        \
      own0[r] = (bf16_t)p0;                                                    \
      own1[r] = (bf16_t)p1;                                                    \
    }                                                                          \
    _Pragma("unroll") for (int t = 0; t < 4; ++t)                              \
      _Pragma("unroll") for (int r = 0; r < 8; ++r) acc[t][r] *= corr;         \
    v4i i0v = __builtin_bit_cast(v4i, own0);                                   \
    v4i i1v = __builtin_bit_cast(v4i, own1);                                   \
    v4i p0v, p1v;                                                              \
    _Pragma("unroll") for (int w = 0; w < 4; ++w) {                            \
      p0v[w] = __shfl_xor(i0v[w], 16, 32);                                     \
      p1v[w] = __shfl_xor(i1v[w], 16, 32);                                     \
    }                                                                          \
    v8bf part0 = __builtin_bit_cast(v8bf, p0v);                                \
    v8bf part1 = __builtin_bit_cast(v8bf, p1v);                                \
    v8bf blo = g ? part1 : own0;                                               \
    v8bf bhi = g ? own1 : part0;                                               \
    v16bf pfrag = __builtin_shufflevector(                                     \
        blo, bhi, 0,1,2,3,4,5,6,7,8,9,10,11,12,13,14,15);                      \
    acc[0] = WMMA_BF16(vf0, pfrag, acc[0]);                                    \
    acc[1] = WMMA_BF16(vf1, pfrag, acc[1]);                                    \
    acc[2] = WMMA_BF16(vf2, pfrag, acc[2]);                                    \
    acc[3] = WMMA_BF16(vf3, pfrag, acc[3]);                                    \
  }

__global__ __launch_bounds__(128) void attention_kernel(const bf16_t* __restrict__ Qb,
                                                        const bf16_t* __restrict__ Kg,
                                                        const bf16_t* __restrict__ VgT,
                                                        bf16_t* __restrict__ AO) {
  const int E = 1024, Tq = 1024, Mg = 512;
  const float scaling = 0.125f;  // 64^-0.5
  int lane = threadIdx.x & 31;
  int wave = threadIdx.x >> 5;
  int b = blockIdx.z, h = blockIdx.y;
  int q0 = blockIdx.x * 64 + wave * 16;
  int g = lane >> 4, l15 = lane & 15;

  const bf16_t* qbase = Qb + ((size_t)b * Tq) * E + h * 64;
  const bf16_t* kbase = Kg + (((size_t)b * 16 + h) * Mg) * 64;
  const bf16_t* vbase = VgT + (((size_t)b * 16 + h) * 64) * Mg;

  // Q^T B-fragments over d = 0..63 (two K-halves), hoisted out of the key loop
  v16bf qf0 = load_bfragT(qbase, E, q0, 0, lane);
  v16bf qf1 = load_bfragT(qbase, E, q0, 32, lane);

  v8f acc[4] = {};              // out^T: 64 d x 16 q
  float m_run = -1e30f, l_run = 0.f;

  v16bf kfA[4], kfB[4];
  kfA[0] = load_afrag(kbase, 64, 0, 0, lane);
  kfA[1] = load_afrag(kbase, 64, 0, 32, lane);
  kfA[2] = load_afrag(kbase, 64, 16, 0, lane);
  kfA[3] = load_afrag(kbase, 64, 16, 32, lane);

  for (int jc = 0; jc < Mg; jc += 64) {
    ATTN_CHUNK(kfA, kfB, jc, jc + 32);
    int jn = (jc + 64 < Mg) ? (jc + 64) : 0;  // wrap: harmless redundant prefetch at end
    ATTN_CHUNK(kfB, kfA, jc + 32, jn);
  }

  float l_full = l_run + __shfl_xor(l_run, 16, 32);
  float inv = 1.0f / l_full;
  bf16_t* obase = AO + ((size_t)(b * Tq) + q0 + l15) * E + h * 64;
#pragma unroll
  for (int t = 0; t < 4; ++t) {
    v8bf ob;
#pragma unroll
    for (int r = 0; r < 8; ++r) ob[r] = (bf16_t)(acc[t][r] * inv);
    *(v8bf*)(obase + t * 16 + g * 8) = ob;
  }
}

// ---------------- host ----------------
extern "C" void kernel_launch(void* const* d_in, const int* in_sizes, int n_in,
                              void* d_out, int out_size, void* d_ws, size_t ws_size,
                              hipStream_t stream) {
  const int B = 2, Tq = 1024, Tk = 8192, E = 1024, H = 16, Mg = 512;
  (void)in_sizes; (void)n_in; (void)out_size; (void)ws_size; (void)Tk;

  const float* hidden = (const float*)d_in[0];
  const float* kv     = (const float*)d_in[1];
  const float* Wq     = (const float*)d_in[2];
  const float* bq     = (const float*)d_in[3];
  const float* Wk     = (const float*)d_in[4];
  const float* bk     = (const float*)d_in[5];
  const float* Wv     = (const float*)d_in[6];
  const float* bv     = (const float*)d_in[7];
  const float* Wo     = (const float*)d_in[8];
  const float* bo     = (const float*)d_in[9];

  char* ws = (char*)d_ws;
  size_t off = 0;
  auto carve = [&](size_t bytes) {
    void* p = ws + off;
    off += (bytes + 255) & ~(size_t)255;
    return p;
  };
  bf16_t* hb   = (bf16_t*)carve((size_t)B * Tq * E * 2);   // hidden bf16
  bf16_t* Wqb  = (bf16_t*)carve((size_t)E * E * 2);
  bf16_t* Wkb  = (bf16_t*)carve((size_t)E * E * 2);
  bf16_t* Wvb  = (bf16_t*)carve((size_t)E * E * 2);
  bf16_t* Wob  = (bf16_t*)carve((size_t)E * E * 2);
  bf16_t* Qb   = (bf16_t*)carve((size_t)B * Tq * E * 2);
  bf16_t* Kgb  = (bf16_t*)carve((size_t)B * H * Mg * 64 * 2);
  bf16_t* VgTb = (bf16_t*)carve((size_t)B * H * 64 * Mg * 2);
  bf16_t* AOb  = (bf16_t*)carve((size_t)B * Tq * E * 2);

  // 0) fp32 -> bf16 converts
  {
    int n4 = B * Tq * E / 4;
    cvt_f32_bf16_kernel<<<(n4 + 255) / 256, 256, 0, stream>>>(hidden, hb, n4);
    int w4 = E * E / 4;
    cvt_f32_bf16_kernel<<<(w4 + 255) / 256, 256, 0, stream>>>(Wq, Wqb, w4);
    cvt_f32_bf16_kernel<<<(w4 + 255) / 256, 256, 0, stream>>>(Wk, Wkb, w4);
    cvt_f32_bf16_kernel<<<(w4 + 255) / 256, 256, 0, stream>>>(Wv, Wvb, w4);
    cvt_f32_bf16_kernel<<<(w4 + 255) / 256, 256, 0, stream>>>(Wo, Wob, w4);
  }
  // 1) Q = hidden @ Wq^T + bq  (bf16 out, token-major [B*Tq][E])
  gemm_xwt_kernel<<<dim3(B * Tq / 16, E / 256), 128, 0, stream>>>(hb, Wqb, bq, Qb, nullptr, E, E);
  // 2) gathered K/V projection (only head t%16 per token)
  kv_gather_proj_kernel<<<dim3(Mg / 64, H, B), 256, 0, stream>>>(kv, Wkb, bk, Wvb, bv, Kgb, VgTb);
  // 3) attention
  attention_kernel<<<dim3(Tq / 64, H, B), 128, 0, stream>>>(Qb, Kgb, VgTb, AOb);
  // 4) out = AO @ Wo^T + bo (fp32 out)
  gemm_xwt_kernel<<<dim3(B * Tq / 16, E / 256), 128, 0, stream>>>(AOb, Wob, bo, nullptr, (float*)d_out, E, E);
}
